// UpdateEdge_24927990186026
// MI455X (gfx1250) — compile-verified, matrix operand order checked by reading, not run on hardware
//
#include <hip/hip_runtime.h>
#include <hip/hip_bf16.h>
#include <stdint.h>

// ---------------------------------------------------------------------------
// Allegro-style edge update for MI455X (gfx1250, wave32, WMMA).
// Dense algebra: v_wmma_f32_16x16x32_bf16 (bf16 A/B, f32 acc).
// A-tile staging: GLOBAL_LOAD_ASYNC_TO_LDS_B128 + ASYNCcnt double-buffering
// when the toolchain exposes the builtin; synchronous LDS staging otherwise.
// ---------------------------------------------------------------------------

#define NSC   128
#define NVC   64
#define DIMF  320
#define LATD  128
#define OHD   128

#define CO_CONST 0.8944271909999159f  // 1/sqrt(1.25)
#define CN_CONST 0.4472135954999579f  // 0.5/sqrt(1.25)

typedef __attribute__((ext_vector_type(16))) __bf16 v16bf;
typedef __attribute__((ext_vector_type(8)))  float  v8f;
typedef __attribute__((ext_vector_type(4)))  int    v4i;

#if defined(__HIP_DEVICE_COMPILE__) && defined(__gfx1250__) && \
    __has_builtin(__builtin_amdgcn_global_load_async_to_lds_b128)
#define USE_ASYNC_LDS 1
#else
#define USE_ASYNC_LDS 0
#endif

__device__ __forceinline__ float sigmoidf_(float x) { return 1.0f / (1.0f + __expf(-x)); }
__device__ __forceinline__ float siluf_(float x)    { return x * sigmoidf_(x); }

__device__ __forceinline__ void wait_async_le1() {
#if USE_ASYNC_LDS
#if __has_builtin(__builtin_amdgcn_s_wait_asynccnt)
  __builtin_amdgcn_s_wait_asynccnt(1);
#else
  asm volatile("s_wait_asynccnt 0x1" ::: "memory");
#endif
#endif
}
__device__ __forceinline__ void wait_async_le0() {
#if USE_ASYNC_LDS
#if __has_builtin(__builtin_amdgcn_s_wait_asynccnt)
  __builtin_amdgcn_s_wait_asynccnt(0);
#else
  asm volatile("s_wait_asynccnt 0x0" ::: "memory");
#endif
#endif
}

__device__ __forceinline__ v16bf f32x16_to_bf16(float4 a0, float4 a1, float4 a2, float4 a3) {
  v16bf r;
  r[0]  = (__bf16)a0.x; r[1]  = (__bf16)a0.y; r[2]  = (__bf16)a0.z; r[3]  = (__bf16)a0.w;
  r[4]  = (__bf16)a1.x; r[5]  = (__bf16)a1.y; r[6]  = (__bf16)a1.z; r[7]  = (__bf16)a1.w;
  r[8]  = (__bf16)a2.x; r[9]  = (__bf16)a2.y; r[10] = (__bf16)a2.z; r[11] = (__bf16)a2.w;
  r[12] = (__bf16)a3.x; r[13] = (__bf16)a3.y; r[14] = (__bf16)a3.z; r[15] = (__bf16)a3.w;
  return r;
}

// ---------------------------------------------------------------------------
// Generic WMMA GEMM:  C[M,N] = act( alpha * A[M,K] @ W[N,K]^T + bias + Cacc )
//   block = 128 threads (4 waves, wave32); block tile = 16 rows x 64 cols,
//   each wave owns one 16x16 WMMA tile. A tile (16x32 f32) is double-buffered
//   in LDS via async-to-LDS copies; f32->bf16 conversion happens on fragment
//   read (VALU co-executes with the XDL WMMA pipe).
//   rowidx != nullptr gathers A rows. act: 0 = none, 1 = SiLU.
// Requires: K % 32 == 0, N % 64 == 0 (true for every GEMM here).
// ---------------------------------------------------------------------------
__global__ __launch_bounds__(128, 1)
void gemm_wmma_bf16(const float* __restrict__ A, int lda, const int* __restrict__ rowidx,
                    const float* __restrict__ W, int ldw,
                    float* __restrict__ C, int ldc,
                    const float* __restrict__ Cacc,
                    const float* __restrict__ bias,
                    int M, int N, int K, float alpha, int act) {
  // 36-float rows: 144B stride -> 16B aligned b128 rows, bank-spread across 64 banks
  __shared__ float As[2][16][36];

  const int tid   = threadIdx.x;
  const int lane  = tid & 31;
  const int wid   = tid >> 5;
  const int rowb  = blockIdx.x * 16;
  const int colb  = blockIdx.y * 64 + wid * 16;

  const int khalf = lane >> 4;                  // 0: lanes 0-15, 1: lanes 16-31
  const int mn    = lane & 15;

  // cooperative A-tile loader: 128 thr * 16B = 16 rows x 32 floats
  const int ar = tid >> 3;                      // 0..15 (row in tile)
  const int ac = (tid & 7) * 4;                 // 0..28 (col in tile)
  int lrow = rowb + ar; if (lrow >= M) lrow = M - 1;
  const int prow = rowidx ? rowidx[lrow] : lrow;
  const float* __restrict__ Arow = A + (size_t)prow * lda;

  const int ncol = colb + mn;
  const float4* __restrict__ Wrow = reinterpret_cast<const float4*>(W + (size_t)ncol * ldw);

  v8f acc = {};

  auto stage = [&](int buf, int kb) {
#if USE_ASYNC_LDS
    const float* g = Arow + kb + ac;
    float* l = &As[buf][ar][ac];
    v4i __attribute__((address_space(1)))* gp =
        (v4i __attribute__((address_space(1)))*)(unsigned long long)(uintptr_t)g;
    v4i __attribute__((address_space(3)))* lp =
        (v4i __attribute__((address_space(3)))*)(unsigned long long)(uintptr_t)l;
    __builtin_amdgcn_global_load_async_to_lds_b128(gp, lp, 0, 0);
#else
    float4 av = *reinterpret_cast<const float4*>(Arow + kb + ac);
    *reinterpret_cast<float4*>(&As[buf][ar][ac]) = av;
#endif
  };

  stage(0, 0);
  int cur = 0;

  for (int kb = 0; kb < K; kb += 32) {
    const bool more = (kb + 32) < K;
    if (more) stage(cur ^ 1, kb + 32);          // overlap next tile's DMA with this WMMA
#if USE_ASYNC_LDS
    if (more) wait_async_le1(); else wait_async_le0();  // in-order: current tile resident
#endif
    __syncthreads();

    // ---- A fragment (16-bit 16x32 layout): elem j<8 -> K=kh*8+j ; j>=8 -> K=16+kh*8+(j-8)
    const float4* ap = reinterpret_cast<const float4*>(&As[cur][mn][0]);
    const int p = khalf * 2;
    v16bf af = f32x16_to_bf16(ap[p], ap[p + 1], ap[p + 4], ap[p + 5]);

    // ---- B fragment straight from global W (row = output col, same K mapping)
    const int q = (kb >> 2) + khalf * 2;
    v16bf bf_ = f32x16_to_bf16(Wrow[q], Wrow[q + 1], Wrow[q + 4], Wrow[q + 5]);

    // prefetch next W K-slab (speculative; dropped if OOB) -> global_prefetch_b8
    __builtin_prefetch(reinterpret_cast<const float*>(Wrow) + kb + 32, 0, 1);

    // D = A x B + C   (f32 accumulate)
    acc = __builtin_amdgcn_wmma_f32_16x16x32_bf16(false, af, false, bf_, (short)0, acc,
                                                  false, false);
    __syncthreads();
    cur ^= 1;
  }

  // ---- epilogue: C/D layout n = lane&15, m = 8*(lane>>4) + r
  const int m0 = khalf * 8;
#pragma unroll
  for (int r = 0; r < 8; ++r) {
    int row = rowb + m0 + r;
    if (row < M && ncol < N) {
      float v = alpha * acc[r];
      if (bias) v += bias[ncol];
      if (Cacc) v += Cacc[(size_t)row * ldc + ncol];
      if (act == 1) v = siluf_(v);
      C[(size_t)row * ldc + ncol] = v;
    }
  }
}

// ---------------------------------------------------------------------------
// Elementwise / gather kernels
// ---------------------------------------------------------------------------

// rotation-to-z per edge (Rodrigues, with az ~ -1 fallback); writes R row-major
__global__ void k_rot(const float* __restrict__ evec, const int* __restrict__ act,
                      float* __restrict__ Rout, int E) {
  int e = blockIdx.x * blockDim.x + threadIdx.x;
  if (e >= E) return;
  int ae = act[e];
  float x = evec[3 * (size_t)ae], y = evec[3 * (size_t)ae + 1], z = evec[3 * (size_t)ae + 2];
  float n = sqrtf(x * x + y * y + z * z) + 1e-12f;
  float ax = x / n, ay = y / n, az = z / n;
  float* Rm = Rout + (size_t)e * 9;
  if (1.0f + az < 1e-6f) {
    Rm[0] = 1.f; Rm[1] = 0.f;  Rm[2] = 0.f;
    Rm[3] = 0.f; Rm[4] = -1.f; Rm[5] = 0.f;
    Rm[6] = 0.f; Rm[7] = 0.f;  Rm[8] = -1.f;
  } else {
    float ic = 1.0f / (1.0f + az);
    Rm[0] = 1.f - ax * ax * ic; Rm[1] = -ax * ay * ic;      Rm[2] = -ax;
    Rm[3] = -ax * ay * ic;      Rm[4] = 1.f - ay * ay * ic; Rm[5] = -ay;
    Rm[6] = ax;                 Rm[7] = ay;                 Rm[8] = 1.f - (ax * ax + ay * ay) * ic;
  }
}

// in_s[e,0:384] = [nf[src,:128], ef[e,:128], nf[dst,:128]]
__global__ void k_build_ins(const float* __restrict__ nf, const float* __restrict__ ef,
                            const int* __restrict__ eidx, const int* __restrict__ act,
                            float* __restrict__ ins, int E, int NEi) {
  int t = blockIdx.x * blockDim.x + threadIdx.x;
  if (t >= E * 384) return;
  int e = t / 384, k = t % 384;
  int sel = k >> 7, kk = k & 127;
  int ae = act[e];
  float v;
  if (sel == 0)      v = nf[(size_t)eidx[ae] * DIMF + kk];
  else if (sel == 1) v = ef[(size_t)e * DIMF + kk];
  else               v = nf[(size_t)eidx[NEi + ae] * DIMF + kk];
  ins[t] = v;
}

// rotated vector channels: vr_i[e,c] = sum_j R[e,i,j] * in_v[e,c,j]
__global__ void k_build_vr(const float* __restrict__ nf, const float* __restrict__ ef,
                           const int* __restrict__ eidx, const int* __restrict__ act,
                           const float* __restrict__ R,
                           float* __restrict__ vr0, float* __restrict__ vr1,
                           float* __restrict__ vr2, int E, int NEi) {
  int t = blockIdx.x * blockDim.x + threadIdx.x;
  if (t >= E * 192) return;
  int e = t / 192, c = t % 192;
  int g = c >> 6, cc = c & 63;
  int ae = act[e];
  const float* src;
  if (g == 0)      src = nf + (size_t)eidx[ae] * DIMF;
  else if (g == 1) src = ef + (size_t)e * DIMF;
  else             src = nf + (size_t)eidx[NEi + ae] * DIMF;
  float v0 = src[NSC + 3 * cc], v1 = src[NSC + 3 * cc + 1], v2 = src[NSC + 3 * cc + 2];
  const float* Rm = R + (size_t)e * 9;
  vr0[t] = Rm[0] * v0 + Rm[1] * v1 + Rm[2] * v2;
  vr1[t] = Rm[3] * v0 + Rm[4] * v1 + Rm[5] * v2;
  vr2[t] = Rm[6] * v0 + Rm[7] * v1 + Rm[8] * v2;
}

// out_v = R^T @ [xo,yo,zo], gated by sigmoid(out_s[:,128:])
__global__ void k_unrotate_gate(const float* __restrict__ xo, const float* __restrict__ yo,
                                const float* __restrict__ zo, const float* __restrict__ R,
                                const float* __restrict__ outs,
                                float* __restrict__ vx, float* __restrict__ vy,
                                float* __restrict__ vz, int E) {
  int t = blockIdx.x * blockDim.x + threadIdx.x;
  if (t >= E * 64) return;
  int e = t >> 6, c = t & 63;
  float o0 = xo[t], o1 = yo[t], o2 = zo[t];
  const float* Rm = R + (size_t)e * 9;
  float g = sigmoidf_(outs[(size_t)e * 192 + NSC + c]);
  vx[t] = (Rm[0] * o0 + Rm[3] * o1 + Rm[6] * o2) * g;
  vy[t] = (Rm[1] * o0 + Rm[4] * o1 + Rm[7] * o2) * g;
  vz[t] = (Rm[2] * o0 + Rm[5] * o1 + Rm[8] * o2) * g;
}

__global__ void k_silu_s(const float* __restrict__ outs, float* __restrict__ s, int E) {
  int t = blockIdx.x * blockDim.x + threadIdx.x;
  if (t >= E * 128) return;
  s[t] = siluf_(outs[(size_t)(t >> 7) * 192 + (t & 127)]);
}

// LayerNorm of latents[active_edges] -> hbuf1[:, :128]; one wave32 per edge
__global__ void k_layernorm(const float* __restrict__ lat, const int* __restrict__ act,
                            const float* __restrict__ g, const float* __restrict__ b,
                            float* __restrict__ hb1, int E) {
  int e = blockIdx.x;
  if (e >= E) return;
  int lane = threadIdx.x;  // 32
  const float* row = lat + (size_t)act[e] * LATD;
  float4 v = reinterpret_cast<const float4*>(row)[lane];
  float s = v.x + v.y + v.z + v.w;
  float s2 = v.x * v.x + v.y * v.y + v.z * v.z + v.w * v.w;
#pragma unroll
  for (int off = 16; off; off >>= 1) {
    s  += __shfl_xor(s, off, 32);
    s2 += __shfl_xor(s2, off, 32);
  }
  float mu  = s * (1.0f / 128.0f);
  float var = s2 * (1.0f / 128.0f) - mu * mu;
  float inv = rsqrtf(var + 1e-5f);
  float* out = hb1 + (size_t)e * 256 + lane * 4;
  out[0] = (v.x - mu) * inv * g[lane * 4 + 0] + b[lane * 4 + 0];
  out[1] = (v.y - mu) * inv * g[lane * 4 + 1] + b[lane * 4 + 1];
  out[2] = (v.z - mu) * inv * g[lane * 4 + 2] + b[lane * 4 + 2];
  out[3] = (v.w - mu) * inv * g[lane * 4 + 3] + b[lane * 4 + 3];
}

__global__ void k_copy_onehot(const float* __restrict__ oh, float* __restrict__ hb2, int E) {
  int t = blockIdx.x * blockDim.x + threadIdx.x;
  if (t >= E * 128) return;
  hb2[(size_t)(t >> 7) * 256 + 128 + (t & 127)] = oh[t];
}

// de-interleave edge_features vector part into x/y/z planes
__global__ void k_extract_rv(const float* __restrict__ ef, float* __restrict__ rvx,
                             float* __restrict__ rvy, float* __restrict__ rvz, int E) {
  int t = blockIdx.x * blockDim.x + threadIdx.x;
  if (t >= E * 64) return;
  const float* base = ef + (size_t)(t >> 6) * DIMF + NSC + 3 * (t & 63);
  rvx[t] = base[0]; rvy[t] = base[1]; rvz[t] = base[2];
}

// es = (cn*(scalars*w_s) + co*rs) * (1 + w0)  -> ef[:, :128]
__global__ void k_ef_s(const float* __restrict__ hb1, const float* __restrict__ wbuf,
                       const float* __restrict__ rs, const float* __restrict__ w0b,
                       float* __restrict__ efo, int E) {
  int t = blockIdx.x * blockDim.x + threadIdx.x;
  if (t >= E * 128) return;
  int e = t >> 7, k = t & 127;
  float sc = hb1[(size_t)e * 256 + 128 + k];
  float sv = sc * wbuf[(size_t)e * 192 + k];
  float es = CN_CONST * sv + CO_CONST * rs[t];
  es *= 1.0f + w0b[t];
  efo[(size_t)e * DIMF + k] = es;
}

// ev = (cn*(v2*w_v) + co*rv2) * (1 + w1), re-interleave -> ef[:, 128+3c+i]
__global__ void k_ef_v(const float* __restrict__ vx2, const float* __restrict__ vy2,
                       const float* __restrict__ vz2, const float* __restrict__ wbuf,
                       const float* __restrict__ rvx2, const float* __restrict__ rvy2,
                       const float* __restrict__ rvz2, const float* __restrict__ w1b,
                       float* __restrict__ efo, int E) {
  int t = blockIdx.x * blockDim.x + threadIdx.x;
  if (t >= E * 64) return;
  int e = t >> 6, c = t & 63;
  float wv = wbuf[(size_t)e * 192 + 128 + c];
  float wm = 1.0f + w1b[t];
  float* o = efo + (size_t)e * DIMF + NSC + 3 * c;
  o[0] = (CN_CONST * vx2[t] * wv + CO_CONST * rvx2[t]) * wm;
  o[1] = (CN_CONST * vy2[t] * wv + CO_CONST * rvy2[t]) * wm;
  o[2] = (CN_CONST * vz2[t] * wv + CO_CONST * rvz2[t]) * wm;
}

__global__ void k_copy_lat(const float* __restrict__ lat, float* __restrict__ lo, int n) {
  int t = blockIdx.x * blockDim.x + threadIdx.x;
  if (t < n) lo[t] = lat[t];
}

// lat_out[active[e]] = cn*cutoff[active[e]]*nl2[e] + co*latents[active[e]]
__global__ void k_lat_scatter(const float* __restrict__ lat, const int* __restrict__ act,
                              const float* __restrict__ cut, const float* __restrict__ nl2,
                              float* __restrict__ lo, int E) {
  int t = blockIdx.x * blockDim.x + threadIdx.x;
  if (t >= E * 128) return;
  int e = t >> 7, k = t & 127;
  int row = act[e];
  float nlv = cut[row] * nl2[t];
  lo[(size_t)row * LATD + k] = CN_CONST * nlv + CO_CONST * lat[(size_t)row * LATD + k];
}

// ---------------------------------------------------------------------------
extern "C" void kernel_launch(void* const* d_in, const int* in_sizes, int n_in,
                              void* d_out, int out_size, void* d_ws, size_t ws_size,
                              hipStream_t stream) {
  const float* latents = (const float*)d_in[0];
  const float* node_f  = (const float*)d_in[1];
  const float* edge_f  = (const float*)d_in[3];
  const int*   eidx    = (const int*)d_in[4];
  const float* evec    = (const float*)d_in[5];
  const float* cutoff  = (const float*)d_in[6];
  const int*   act_e   = (const int*)d_in[7];
  const float* onehot  = (const float*)d_in[8];
  const float* W_s     = (const float*)d_in[10];
  const float* W0      = (const float*)d_in[11];
  const float* Wr      = (const float*)d_in[12];
  const float* Wi      = (const float*)d_in[13];
  const float* Wl0     = (const float*)d_in[14];
  const float* bl0     = (const float*)d_in[15];
  const float* Wl1     = (const float*)d_in[16];
  const float* Wres0   = (const float*)d_in[17];
  const float* bres0   = (const float*)d_in[18];
  const float* Wres1   = (const float*)d_in[19];
  const float* W_embed = (const float*)d_in[20];
  const float* W1a     = (const float*)d_in[21];
  const float* W1b     = (const float*)d_in[22];
  const float* W1c     = (const float*)d_in[23];
  const float* W2a     = (const float*)d_in[24];
  const float* W2b     = (const float*)d_in[25];
  const float* W2c     = (const float*)d_in[26];
  const float* ln_g    = (const float*)d_in[27];
  const float* ln_b    = (const float*)d_in[28];
  const float* Wtp0    = (const float*)d_in[29];
  const float* Wtp1    = (const float*)d_in[30];

  const int E   = in_sizes[7];            // active edges
  const int NEt = in_sizes[0] / LATD;     // total edges (latents rows)
  const int NEi = in_sizes[4] / 2;        // edge_index second-row offset
  const size_t Es = (size_t)E;

  // output layout: ef [E,320] | lat_out [NEt,128] | R [E,3,3]
  float* ef_out  = (float*)d_out;
  float* lat_out = ef_out + Es * DIMF;
  float* R_out   = lat_out + (size_t)NEt * LATD;

  // workspace (floats). Phase-2 buffers alias phase-1 (dead after unrotate).
  float* ws = (float*)d_ws;
  // phase 1
  float* in_s = ws;                 // 384E
  float* vr0  = ws + 384 * Es;      // 192E
  float* vr1  = ws + 576 * Es;
  float* vr2  = ws + 768 * Es;
  float* outs = ws + 960 * Es;      // 192E
  float* xo   = ws + 1152 * Es;     // 64E
  float* yo   = ws + 1216 * Es;
  float* zo   = ws + 1280 * Es;     // phase-1 end: 1344E
  // phase 2 (aliases phase 1)
  float* hb1  = ws;                 // 256E : [xln | scalars]
  float* hb2  = ws + 256 * Es;      // 256E : [nl | onehot]
  float* wbuf = ws + 512 * Es;      // 192E
  float* g1   = ws + 704 * Es;      // 128E
  float* g2   = ws + 832 * Es;      // 128E
  float* nl2  = ws + 960 * Es;      // 128E
  float* rs   = ws + 1088 * Es;     // 128E
  float* rvx  = ws + 1216 * Es;     // 64E x3
  float* rvy  = ws + 1280 * Es;
  float* rvz  = ws + 1344 * Es;
  float* rvx2 = ws + 1408 * Es;     // 64E x3
  float* rvy2 = ws + 1472 * Es;
  float* rvz2 = ws + 1536 * Es;
  float* w0b  = ws + 1600 * Es;     // 128E
  float* w1b  = ws + 1728 * Es;     // 64E  -> shared region = 1792E
  // persistent
  float* P    = ws + 1792 * Es;
  float* s_act = P;                 // 128E
  float* vx   = P + 128 * Es;       // 64E x3 (gated, rotated-back)
  float* vy   = P + 192 * Es;
  float* vz   = P + 256 * Es;
  float* vx2  = P + 320 * Es;       // 64E x3 (after Wl1)
  float* vy2  = P + 384 * Es;
  float* vz2  = P + 448 * Es;       // total ws = 2304E floats

  const float KS1 = 0.05103103630798288f;   // 1/sqrt(384)
  const float KV  = 0.07216878364870323f;   // 1/sqrt(192)
  const float KL  = 0.08838834764831845f;   // 1/sqrt(128)
  const float K64 = 0.125f;                 // 1/sqrt(64)
  const float K256 = 0.0625f;               // 1/sqrt(256)

  auto cdiv = [](int a, int b) { return (a + b - 1) / b; };
#define GEMM(Ap, lda_, ridx, Wp, ldw_, Cp, ldc_, Cacc_, bias_, N_, K_, al_, act_)            \
  gemm_wmma_bf16<<<dim3(cdiv(E, 16), (N_) / 64), dim3(128), 0, stream>>>(                    \
      Ap, lda_, ridx, Wp, ldw_, Cp, ldc_, Cacc_, bias_, E, N_, K_, al_, act_)

  // ---- phase 1: gather + rotate ----
  k_rot<<<cdiv(E, 256), 256, 0, stream>>>(evec, act_e, R_out, E);
  k_build_ins<<<cdiv(E * 384, 256), 256, 0, stream>>>(node_f, edge_f, eidx, act_e, in_s, E, NEi);
  k_build_vr<<<cdiv(E * 192, 256), 256, 0, stream>>>(node_f, edge_f, eidx, act_e, R_out,
                                                     vr0, vr1, vr2, E, NEi);

  GEMM(in_s, 384, nullptr, W_s, 384, outs, 192, nullptr, nullptr, 192, 384, KS1, 0);
  GEMM(vr0, 192, nullptr, Wr, 192, xo, 64, nullptr, nullptr, 64, 192,  KV, 0);
  GEMM(vr1, 192, nullptr, Wi, 192, xo, 64, xo,      nullptr, 64, 192, -KV, 0);
  GEMM(vr0, 192, nullptr, Wi, 192, yo, 64, nullptr, nullptr, 64, 192,  KV, 0);
  GEMM(vr1, 192, nullptr, Wr, 192, yo, 64, yo,      nullptr, 64, 192,  KV, 0);
  GEMM(vr2, 192, nullptr, W0, 192, zo, 64, nullptr, nullptr, 64, 192,  KV, 0);

  k_unrotate_gate<<<cdiv(E * 64, 256), 256, 0, stream>>>(xo, yo, zo, R_out, outs, vx, vy, vz, E);
  k_silu_s<<<cdiv(E * 128, 256), 256, 0, stream>>>(outs, s_act, E);
  // phase-1 scratch (in_s, vr*, outs, xo/yo/zo) dead from here on.

  // ---- phase 2 ----
  GEMM(s_act, 128, nullptr, Wl0, 128, hb1 + 128, 256, nullptr, bl0, 128, 128, KL, 0); // scalars
  GEMM(vx, 64, nullptr, Wl1, 64, vx2, 64, nullptr, nullptr, 64, 64, K64, 0);
  GEMM(vy, 64, nullptr, Wl1, 64, vy2, 64, nullptr, nullptr, 64, 64, K64, 0);
  GEMM(vz, 64, nullptr, Wl1, 64, vz2, 64, nullptr, nullptr, 64, 64, K64, 0);

  k_layernorm<<<E, 32, 0, stream>>>(latents, act_e, ln_g, ln_b, hb1, E);          // xln
  GEMM(latents, 128, act_e, W_embed, 128, wbuf, 192, nullptr, nullptr, 192, 128, KL, 0); // w

  GEMM(hb1, 256, nullptr, W1a, 256, g1, 128, nullptr, nullptr, 128, 256, K256, 1);
  GEMM(g1, 128, nullptr, W1b, 128, g2, 128, nullptr, nullptr, 128, 128, KL, 1);
  GEMM(g2, 128, nullptr, W1c, 128, hb2, 256, nullptr, nullptr, 128, 128, KL, 0);  // nl
  k_copy_onehot<<<cdiv(E * 128, 256), 256, 0, stream>>>(onehot, hb2, E);

  GEMM(hb2, 256, nullptr, W2a, 256, g1, 128, nullptr, nullptr, 128, 256, K256, 1);
  GEMM(g1, 128, nullptr, W2b, 128, g2, 128, nullptr, nullptr, 128, 128, KL, 1);
  GEMM(g2, 128, nullptr, W2c, 128, nl2, 128, nullptr, nullptr, 128, 128, KL, 0);

  GEMM(edge_f, DIMF, nullptr, Wres0, 128, rs, 128, nullptr, bres0, 128, 128, KL, 0);
  k_extract_rv<<<cdiv(E * 64, 256), 256, 0, stream>>>(edge_f, rvx, rvy, rvz, E);
  GEMM(rvx, 64, nullptr, Wres1, 64, rvx2, 64, nullptr, nullptr, 64, 64, K64, 0);
  GEMM(rvy, 64, nullptr, Wres1, 64, rvy2, 64, nullptr, nullptr, 64, 64, K64, 0);
  GEMM(rvz, 64, nullptr, Wres1, 64, rvz2, 64, nullptr, nullptr, 64, 64, K64, 0);

  GEMM(onehot, 128, nullptr, Wtp0, 128, w0b, 128, nullptr, nullptr, 128, 128, KL, 0);
  GEMM(onehot, 128, nullptr, Wtp1, 128, w1b, 64,  nullptr, nullptr, 64,  128, KL, 0);

  // ---- final combines ----
  k_ef_s<<<cdiv(E * 128, 256), 256, 0, stream>>>(hb1, wbuf, rs, w0b, ef_out, E);
  k_ef_v<<<cdiv(E * 64, 256), 256, 0, stream>>>(vx2, vy2, vz2, wbuf, rvx2, rvy2, rvz2, w1b,
                                                ef_out, E);
  k_copy_lat<<<cdiv(NEt * 128, 256), 256, 0, stream>>>(latents, lat_out, NEt * 128);
  k_lat_scatter<<<cdiv(E * 128, 256), 256, 0, stream>>>(latents, act_e, cutoff, nl2, lat_out, E);
#undef GEMM
}
